// VRNN_36421322670647
// MI455X (gfx1250) — matrix operand order repaired and checked by reading, not compiled, not run on hardware
//
#include <hip/hip_runtime.h>
#include <hip/hip_bf16.h>

#define TT  64
#define BB  256
#define XD  784
#define XP  832      // XD zero-padded to multiple of 64 for the pipelined WMMA K loop
#define HD  2048
#define ZD  128
#define HD2 4096
#define HD3 6144

typedef __attribute__((ext_vector_type(16))) __bf16 v16bf;
typedef __attribute__((ext_vector_type(8)))  float  v8f;

union FragBF {
    v16bf bf;
    uint4 u[2];
};

// ---------------------------------------------------------------------------
// Activation: 0=none 1=relu 2=sigmoid 3=softplus (stable, matches jax.nn)
// ---------------------------------------------------------------------------
template <int ACT>
__device__ __forceinline__ float apply_act(float v) {
    if (ACT == 1) return v > 0.f ? v : 0.f;
    if (ACT == 2) return 1.f / (1.f + expf(-v));
    if (ACT == 3) return fmaxf(v, 0.f) + log1pf(expf(-fabsf(v)));
    return v;
}

// ---------------------------------------------------------------------------
// bf16 WMMA GEMM:  C[M,N] = act( A[M,K](bf16) @ W[N,K]^T(bf16) + bias[N] )
// W kept in torch layout [N,K] row-major -> B-fragment lane (column n) reads
// two runs of 8 consecutive K bf16 (two b128 loads, 16 K apart).
// Block tile 128x128, 8 waves, each wave a 32x64 tile (8 WMMA per K-step).
// Software pipeline: two STATICALLY-NAMED fragment sets (X/Y) double-buffer
// the K loop (no runtime-indexed arrays -> no scratch spills), peeled tail.
// Requires: M % 128 == 0, K % 64 == 0. N arbitrary (clamped/guarded).
// ---------------------------------------------------------------------------
template <int ACT>
__global__ __launch_bounds__(256) void gemm_kernel(
    const __hip_bfloat16* __restrict__ A,
    const __hip_bfloat16* __restrict__ W,
    const float* __restrict__ bias,
    float* __restrict__ Cf,            // optional fp32 output (or null)
    __hip_bfloat16* __restrict__ Cb,   // optional bf16 output (or null)
    int M, int N, int K)
{
    const int lane = threadIdx.x & 31;
    const int wave = threadIdx.x >> 5;
    const int wm = wave >> 1;                    // 0..3 -> 32-row strip
    const int wn = wave & 1;                     // 0..1 -> 64-col strip
    const int row0 = blockIdx.y * 128 + wm * 32;
    const int col0 = blockIdx.x * 128 + wn * 64;

    v8f c[2][4] = {};

    const int r   = lane & 15;
    const int kgA = (lane >> 4) * 8;             // A frag: lanes 16-31 carry K+8..15 / +24..31
    const int kgB = (lane >> 4) * 8;             // B frag: lanes 16-31 carry K+8..15 / +24..31

    const __hip_bfloat16* pa0 = A + (size_t)(row0 + r) * K + kgA;
    const __hip_bfloat16* pa1 = pa0 + (size_t)16 * K;

    int cc0 = col0 + r;      if (cc0 > N - 1) cc0 = N - 1;   // clamp ragged N
    int cc1 = col0 + 16 + r; if (cc1 > N - 1) cc1 = N - 1;
    int cc2 = col0 + 32 + r; if (cc2 > N - 1) cc2 = N - 1;
    int cc3 = col0 + 48 + r; if (cc3 > N - 1) cc3 = N - 1;
    const __hip_bfloat16* pb0 = W + (size_t)cc0 * K + kgB;
    const __hip_bfloat16* pb1 = W + (size_t)cc1 * K + kgB;
    const __hip_bfloat16* pb2 = W + (size_t)cc2 * K + kgB;
    const __hip_bfloat16* pb3 = W + (size_t)cc3 * K + kgB;

    FragBF aX0, aX1, bX0, bX1, bX2, bX3;         // pipeline stage X
    FragBF aY0, aY1, bY0, bY1, bY2, bY3;         // pipeline stage Y

#define LOAD_SET(A0, A1, B0, B1, B2, B3, off)                                  \
    do {                                                                       \
        A0.u[0] = *(const uint4*)(pa0 + (off));                                \
        A0.u[1] = *(const uint4*)(pa0 + (off) + 16);                           \
        A1.u[0] = *(const uint4*)(pa1 + (off));                                \
        A1.u[1] = *(const uint4*)(pa1 + (off) + 16);                           \
        B0.u[0] = *(const uint4*)(pb0 + (off));                                \
        B0.u[1] = *(const uint4*)(pb0 + (off) + 16);                           \
        B1.u[0] = *(const uint4*)(pb1 + (off));                                \
        B1.u[1] = *(const uint4*)(pb1 + (off) + 16);                           \
        B2.u[0] = *(const uint4*)(pb2 + (off));                                \
        B2.u[1] = *(const uint4*)(pb2 + (off) + 16);                           \
        B3.u[0] = *(const uint4*)(pb3 + (off));                                \
        B3.u[1] = *(const uint4*)(pb3 + (off) + 16);                           \
    } while (0)

#define MMA_SET(A0, A1, B0, B1, B2, B3)                                                              \
    do {                                                                                             \
        c[0][0] = __builtin_amdgcn_wmma_f32_16x16x32_bf16(false, A0.bf, false, B0.bf, (short)0, c[0][0], false, false); \
        c[0][1] = __builtin_amdgcn_wmma_f32_16x16x32_bf16(false, A0.bf, false, B1.bf, (short)0, c[0][1], false, false); \
        c[0][2] = __builtin_amdgcn_wmma_f32_16x16x32_bf16(false, A0.bf, false, B2.bf, (short)0, c[0][2], false, false); \
        c[0][3] = __builtin_amdgcn_wmma_f32_16x16x32_bf16(false, A0.bf, false, B3.bf, (short)0, c[0][3], false, false); \
        c[1][0] = __builtin_amdgcn_wmma_f32_16x16x32_bf16(false, A1.bf, false, B0.bf, (short)0, c[1][0], false, false); \
        c[1][1] = __builtin_amdgcn_wmma_f32_16x16x32_bf16(false, A1.bf, false, B1.bf, (short)0, c[1][1], false, false); \
        c[1][2] = __builtin_amdgcn_wmma_f32_16x16x32_bf16(false, A1.bf, false, B2.bf, (short)0, c[1][2], false, false); \
        c[1][3] = __builtin_amdgcn_wmma_f32_16x16x32_bf16(false, A1.bf, false, B3.bf, (short)0, c[1][3], false, false); \
    } while (0)

    LOAD_SET(aX0, aX1, bX0, bX1, bX2, bX3, 0);
    int k0 = 0;
    for (; k0 + 64 < K; k0 += 64) {
        LOAD_SET(aY0, aY1, bY0, bY1, bY2, bY3, 32);     // prefetch 2nd half-step
        MMA_SET(aX0, aX1, bX0, bX1, bX2, bX3);
        LOAD_SET(aX0, aX1, bX0, bX1, bX2, bX3, 64);     // prefetch next iteration
        MMA_SET(aY0, aY1, bY0, bY1, bY2, bY3);
        pa0 += 64; pa1 += 64;
        pb0 += 64; pb1 += 64; pb2 += 64; pb3 += 64;
    }
    // peeled tail: exactly two 32-K steps remain (K % 64 == 0)
    LOAD_SET(aY0, aY1, bY0, bY1, bY2, bY3, 32);
    MMA_SET(aX0, aX1, bX0, bX1, bX2, bX3);
    MMA_SET(aY0, aY1, bY0, bY1, bY2, bY3);

#undef LOAD_SET
#undef MMA_SET

    // Epilogue. C/D layout: VGPR rr -> M = rr (lanes 0-15) or 8+rr (lanes 16-31),
    // N = col0 + j*16 + (lane&15).
    const int rowb = (lane >> 4) * 8;
    #pragma unroll
    for (int i = 0; i < 2; ++i) {
        #pragma unroll
        for (int j = 0; j < 4; ++j) {
            const int col = col0 + j * 16 + (lane & 15);
            if (col >= N) continue;
            const float bv = bias[col];
            #pragma unroll
            for (int rr = 0; rr < 8; ++rr) {
                const int row = row0 + i * 16 + rowb + rr;
                float v = apply_act<ACT>(c[i][j][rr] + bv);
                const size_t idx = (size_t)row * N + col;
                if (Cf) Cf[idx] = v;
                if (Cb) Cb[idx] = __float2bfloat16(v);
            }
        }
    }
}

// ---------------------------------------------------------------------------
// fp32 -> bf16 convert with optional zero-pad of trailing columns.
// Vectorized x4 (float4 in, 4x bf16 out); requires scols%4==0 && dcols%4==0.
// ---------------------------------------------------------------------------
__global__ void convert_pad_kernel(const float* __restrict__ src,
                                   __hip_bfloat16* __restrict__ dst,
                                   int rows, int scols, int dcols)
{
    const size_t n4 = (size_t)rows * (dcols >> 2);
    const int sc4 = scols >> 2, dc4 = dcols >> 2;
    for (size_t i = (size_t)blockIdx.x * blockDim.x + threadIdx.x; i < n4;
         i += (size_t)gridDim.x * blockDim.x) {
        const int rr = (int)(i / dc4), c4 = (int)(i % dc4);
        __hip_bfloat16 o[4];
        if (c4 < sc4) {
            const float4 v = *(const float4*)(src + ((size_t)rr * scols + (size_t)c4 * 4));
            o[0] = __float2bfloat16(v.x); o[1] = __float2bfloat16(v.y);
            o[2] = __float2bfloat16(v.z); o[3] = __float2bfloat16(v.w);
        } else {
            o[0] = o[1] = o[2] = o[3] = __float2bfloat16(0.f);
        }
        *(uint2*)(dst + i * 4) = *(const uint2*)o;
    }
}

// out[r] = [ a[r] | b[r] ], vectorized as uint4 (8 bf16); requires cols%8==0
__global__ void concat_kernel(const __hip_bfloat16* __restrict__ a,
                              const __hip_bfloat16* __restrict__ b,
                              __hip_bfloat16* __restrict__ out,
                              int rows, int cols)
{
    const int c8 = cols >> 3;
    const size_t n = (size_t)rows * 2 * c8;
    const uint4* av = (const uint4*)a;
    const uint4* bv = (const uint4*)b;
    uint4* ov = (uint4*)out;
    for (size_t i = (size_t)blockIdx.x * blockDim.x + threadIdx.x; i < n;
         i += (size_t)gridDim.x * blockDim.x) {
        const int rr = (int)(i / (2 * c8)), cc = (int)(i % (2 * c8));
        ov[i] = (cc < c8) ? av[(size_t)rr * c8 + cc]
                          : bv[(size_t)rr * c8 + (cc - c8)];
    }
}

__global__ void init_kernel(float* __restrict__ hf, __hip_bfloat16* __restrict__ hb,
                            float* __restrict__ accum)
{
    const size_t n = (size_t)BB * HD;
    for (size_t i = (size_t)blockIdx.x * blockDim.x + threadIdx.x; i < n;
         i += (size_t)gridDim.x * blockDim.x) {
        hf[i] = 0.f;
        hb[i] = __float2bfloat16(0.f);
    }
    if (blockIdx.x == 0 && threadIdx.x < 2) accum[threadIdx.x] = 0.f;
}

__device__ __forceinline__ float block_reduce(float v) {
    #pragma unroll
    for (int o = 16; o > 0; o >>= 1) v += __shfl_down(v, o, 32);
    __shared__ float s[8];
    const int lane = threadIdx.x & 31, w = threadIdx.x >> 5;
    if (lane == 0) s[w] = v;
    __syncthreads();
    v = 0.f;
    if (w == 0) {
        v = (lane < (int)(blockDim.x >> 5)) ? s[lane] : 0.f;
        #pragma unroll
        for (int o = 4; o > 0; o >>= 1) v += __shfl_down(v, o, 32);
    }
    return v;  // valid on thread 0
}

// z = eps*es + em (bf16 out); accum[0] += 0.5 * sum(KLD unit)
__global__ void latent_kernel(const float* __restrict__ eps,
                              const float* __restrict__ em, const float* __restrict__ es,
                              const float* __restrict__ pm, const float* __restrict__ ps,
                              __hip_bfloat16* __restrict__ z, float* __restrict__ accum)
{
    const int n = BB * ZD;
    const int i = blockIdx.x * blockDim.x + threadIdx.x;
    float local = 0.f;
    if (i < n) {
        const float E = es[i], P = ps[i], Em = em[i], Pm = pm[i];
        z[i] = __float2bfloat16(eps[i] * E + Em);
        const float dr = E / P, dm = (Em - Pm) / P;
        local = dm * dm + dr * dr - 2.f * logf(dr) - 1.f;
    }
    const float s = block_reduce(local);
    if (threadIdx.x == 0) atomicAdd(&accum[0], 0.5f * s);
}

// accum[1] += -sum( x*clip(log(dm),-100) + (1-x)*clip(log1p(-dm),-100) )
__global__ void nll_kernel(const float* __restrict__ x, const float* __restrict__ dm,
                           float* __restrict__ accum)
{
    const int n = BB * XD;
    const int i = blockIdx.x * blockDim.x + threadIdx.x;
    float local = 0.f;
    if (i < n) {
        const float d = dm[i], xv = x[i];
        const float lp = fmaxf(logf(d), -100.f);
        const float lq = fmaxf(log1pf(-d), -100.f);
        local = -(xv * lp + (1.f - xv) * lq);
    }
    const float s = block_reduce(local);
    if (threadIdx.x == 0) atomicAdd(&accum[1], s);
}

// GRUCell combine: gi/gh are [B,3*HD] (r,z,n chunks)
__global__ void gru_kernel(const float* __restrict__ gi, const float* __restrict__ gh,
                           float* __restrict__ hf, __hip_bfloat16* __restrict__ hb)
{
    const int n = BB * HD;
    const int i = blockIdx.x * blockDim.x + threadIdx.x;
    if (i >= n) return;
    const int b = i / HD, j = i % HD;
    const size_t base = (size_t)b * HD3;
    const float ir = gi[base + j],           hr = gh[base + j];
    const float iz = gi[base + HD + j],      hz = gh[base + HD + j];
    const float in_ = gi[base + 2 * HD + j], hn = gh[base + 2 * HD + j];
    const float rg = 1.f / (1.f + expf(-(ir + hr)));
    const float ug = 1.f / (1.f + expf(-(iz + hz)));
    const float ng = tanhf(in_ + rg * hn);
    const float hv = (1.f - ug) * ng + ug * hf[i];
    hf[i] = hv;
    hb[i] = __float2bfloat16(hv);
}

__global__ void finalize_kernel(const float* __restrict__ accum, float* __restrict__ out)
{
    if (threadIdx.x < 2) out[threadIdx.x] = accum[threadIdx.x];
}

// ---------------------------------------------------------------------------
// Host-side orchestration
// ---------------------------------------------------------------------------
static void launch_gemm(int act, const __hip_bfloat16* A, const __hip_bfloat16* W,
                        const float* bias, float* Cf, __hip_bfloat16* Cb,
                        int M, int N, int K, hipStream_t s)
{
    dim3 g((N + 127) / 128, M / 128), b(256);
    switch (act) {
        case 0: gemm_kernel<0><<<g, b, 0, s>>>(A, W, bias, Cf, Cb, M, N, K); break;
        case 1: gemm_kernel<1><<<g, b, 0, s>>>(A, W, bias, Cf, Cb, M, N, K); break;
        case 2: gemm_kernel<2><<<g, b, 0, s>>>(A, W, bias, Cf, Cb, M, N, K); break;
        default: gemm_kernel<3><<<g, b, 0, s>>>(A, W, bias, Cf, Cb, M, N, K); break;
    }
}

extern "C" void kernel_launch(void* const* d_in, const int* in_sizes, int n_in,
                              void* d_out, int out_size, void* d_ws, size_t ws_size,
                              hipStream_t stream)
{
    (void)in_sizes; (void)n_in; (void)out_size; (void)ws_size;
    typedef __hip_bfloat16 bf;

    const float* x   = (const float*)d_in[0];
    const float* eps = (const float*)d_in[1];
    const float* Wsrc[15] = {
        (const float*)d_in[2],  (const float*)d_in[4],  (const float*)d_in[6],
        (const float*)d_in[8],  (const float*)d_in[10], (const float*)d_in[12],
        (const float*)d_in[14], (const float*)d_in[16], (const float*)d_in[18],
        (const float*)d_in[20], (const float*)d_in[22], (const float*)d_in[24],
        (const float*)d_in[26], (const float*)d_in[28], (const float*)d_in[30]};
    const float* bia[15] = {
        (const float*)d_in[3],  (const float*)d_in[5],  (const float*)d_in[7],
        (const float*)d_in[9],  (const float*)d_in[11], (const float*)d_in[13],
        (const float*)d_in[15], (const float*)d_in[17], (const float*)d_in[19],
        (const float*)d_in[21], (const float*)d_in[23], (const float*)d_in[25],
        (const float*)d_in[27], (const float*)d_in[29], (const float*)d_in[31]};
    // idx: 0 W1x 1 W2x 2 Wzp 3 Wpr 4 Wpm 5 Wps 6 We1 7 We2 8 Wem 9 Wes
    //      10 Wd1 11 Wd2 12 Wdm 13 Wih 14 Whh
    const int wrows[15] = {HD, HD, HD, HD, ZD, ZD, HD, HD, ZD, ZD, HD, HD, XD, HD3, HD3};
    const int wscol[15] = {XD, HD, ZD, HD, HD, HD, HD2, HD, HD, HD, HD2, HD, HD, HD2, HD};
    const int wdcol[15] = {XP, HD, ZD, HD, HD, HD, HD2, HD, HD, HD, HD2, HD, HD, HD2, HD};

    // --- carve workspace (weights in bf16 -> ~152MB, L2-resident) ---
    char* p = (char*)d_ws;
    auto alloc = [&](size_t bytes) { void* r = p; p += (bytes + 255) & ~(size_t)255; return r; };

    bf* Wbf[15];
    for (int i = 0; i < 15; ++i)
        Wbf[i] = (bf*)alloc((size_t)wrows[i] * wdcol[i] * sizeof(bf));

    bf*    x_bf    = (bf*)alloc((size_t)BB * XP * sizeof(bf));
    float* h_f     = (float*)alloc((size_t)BB * HD * sizeof(float));
    bf*    h_bf    = (bf*)alloc((size_t)BB * HD * sizeof(bf));
    bf*    t1_bf   = (bf*)alloc((size_t)BB * HD * sizeof(bf));
    bf*    phix_bf = (bf*)alloc((size_t)BB * HD * sizeof(bf));
    bf*    prior_bf= (bf*)alloc((size_t)BB * HD * sizeof(bf));
    bf*    enc_bf  = (bf*)alloc((size_t)BB * HD * sizeof(bf));
    bf*    phiz_bf = (bf*)alloc((size_t)BB * HD * sizeof(bf));
    bf*    dec_bf  = (bf*)alloc((size_t)BB * HD * sizeof(bf));
    float* pm_f    = (float*)alloc((size_t)BB * ZD * sizeof(float));
    float* ps_f    = (float*)alloc((size_t)BB * ZD * sizeof(float));
    float* em_f    = (float*)alloc((size_t)BB * ZD * sizeof(float));
    float* es_f    = (float*)alloc((size_t)BB * ZD * sizeof(float));
    bf*    z_bf    = (bf*)alloc((size_t)BB * ZD * sizeof(bf));
    bf*    cat_bf  = (bf*)alloc((size_t)BB * HD2 * sizeof(bf));
    bf*    catg_bf = (bf*)alloc((size_t)BB * HD2 * sizeof(bf));
    float* dm_f    = (float*)alloc((size_t)BB * XD * sizeof(float));
    float* gi_f    = (float*)alloc((size_t)BB * HD3 * sizeof(float));
    float* gh_f    = (float*)alloc((size_t)BB * HD3 * sizeof(float));
    float* accum   = (float*)alloc(2 * sizeof(float));

    // --- one-time per launch: convert weights fp32 -> bf16 (padded) ---
    for (int i = 0; i < 15; ++i) {
        const size_t n4 = (size_t)wrows[i] * (wdcol[i] >> 2);
        const int blocks = (int)((n4 + 255) / 256);
        convert_pad_kernel<<<min(blocks, 8192), 256, 0, stream>>>(
            Wsrc[i], Wbf[i], wrows[i], wscol[i], wdcol[i]);
    }
    init_kernel<<<2048, 256, 0, stream>>>(h_f, h_bf, accum);

    const dim3 b256(256);
    const int catBlocks = (BB * (HD2 / 8) + 255) / 256;
    // --- the recurrence ---
    for (int t = 0; t < TT; ++t) {
        const float* x_t   = x + (size_t)t * BB * XD;
        const float* eps_t = eps + (size_t)t * BB * ZD;

        // phi_x
        convert_pad_kernel<<<(BB * (XP / 4) + 255) / 256, b256, 0, stream>>>(x_t, x_bf, BB, XD, XP);
        launch_gemm(1, x_bf, Wbf[0], bia[0], nullptr, t1_bf,   BB, HD, XP, stream);
        launch_gemm(1, t1_bf, Wbf[1], bia[1], nullptr, phix_bf, BB, HD, HD, stream);
        // prior trunk + heads
        launch_gemm(1, h_bf, Wbf[3], bia[3], nullptr, prior_bf, BB, HD, HD, stream);
        launch_gemm(0, prior_bf, Wbf[4], bia[4], pm_f, nullptr, BB, ZD, HD, stream);
        launch_gemm(3, prior_bf, Wbf[5], bia[5], ps_f, nullptr, BB, ZD, HD, stream);
        // encoder trunk + heads
        concat_kernel<<<catBlocks, b256, 0, stream>>>(phix_bf, h_bf, cat_bf, BB, HD);
        launch_gemm(1, cat_bf, Wbf[6], bia[6], nullptr, t1_bf,  BB, HD, HD2, stream);
        launch_gemm(1, t1_bf, Wbf[7], bia[7], nullptr, enc_bf,  BB, HD, HD, stream);
        launch_gemm(0, enc_bf, Wbf[8], bia[8], em_f, nullptr,   BB, ZD, HD, stream);
        launch_gemm(3, enc_bf, Wbf[9], bia[9], es_f, nullptr,   BB, ZD, HD, stream);
        // reparameterize + KLD
        latent_kernel<<<(BB * ZD + 255) / 256, b256, 0, stream>>>(eps_t, em_f, es_f, pm_f, ps_f, z_bf, accum);
        // phi_z
        launch_gemm(1, z_bf, Wbf[2], bia[2], nullptr, phiz_bf, BB, HD, ZD, stream);
        // decoder trunk + Bernoulli head
        concat_kernel<<<catBlocks, b256, 0, stream>>>(phiz_bf, h_bf, cat_bf, BB, HD);
        launch_gemm(1, cat_bf, Wbf[10], bia[10], nullptr, t1_bf, BB, HD, HD2, stream);
        launch_gemm(1, t1_bf, Wbf[11], bia[11], nullptr, dec_bf, BB, HD, HD, stream);
        launch_gemm(2, dec_bf, Wbf[12], bia[12], dm_f, nullptr,  BB, XD, HD, stream);
        nll_kernel<<<(BB * XD + 255) / 256, b256, 0, stream>>>(x_t, dm_f, accum);
        // GRU
        concat_kernel<<<catBlocks, b256, 0, stream>>>(phix_bf, phiz_bf, catg_bf, BB, HD);
        launch_gemm(0, catg_bf, Wbf[13], bia[13], gi_f, nullptr, BB, HD3, HD2, stream);
        launch_gemm(0, h_bf,   Wbf[14], bia[14], gh_f, nullptr,  BB, HD3, HD, stream);
        gru_kernel<<<(BB * HD + 255) / 256, b256, 0, stream>>>(gi_f, gh_f, h_f, h_bf);
    }

    finalize_kernel<<<1, 32, 0, stream>>>(accum, (float*)d_out);
}
